// MotifPool_75213467288135
// MI455X (gfx1250) — compile-verified, matrix operand order checked by reading, not run on hardware
//
#include <hip/hip_runtime.h>
#include <hip/hip_bf16.h>

typedef __attribute__((ext_vector_type(16))) _Float16 v16h;
typedef __attribute__((ext_vector_type(8)))  float    v8f;

#define DD      256
#define HEADS   4
#define CDIM    64
#define C2      128
#define NGRAPH  2048

// async copy of 16 bytes: global -> LDS, tracked by ASYNCcnt (CDNA5 ISA)
__device__ __forceinline__ void async_g2lds_b128(unsigned lds_byte_off,
                                                 const void* gaddr) {
  asm volatile("global_load_async_to_lds_b128 %0, %1, off"
               :: "v"(lds_byte_off), "v"((unsigned long long)(uintptr_t)gaddr)
               : "memory");
}
__device__ __forceinline__ void wait_asynccnt0() {
  asm volatile("s_wait_asynccnt 0x0" ::: "memory");
}
__device__ __forceinline__ unsigned lds_off(const void* p) {
  return (unsigned)(uintptr_t)p;   // flat LDS aperture: addr[31:0] = LDS offset
}

// ---------------------------------------------------------------- utilities

__global__ void zero_f32_kernel(float* __restrict__ p, long long n) {
  long long i = (long long)blockIdx.x * blockDim.x + threadIdx.x;
  long long stride = (long long)gridDim.x * blockDim.x;
  for (; i < n; i += stride) p[i] = 0.0f;
}

// Wp [256 (k)][256 (n)] f32  ->  Wp16T [n][k] f16 (column fragments contiguous)
__global__ void cvt_transpose_Wp_kernel(const float* __restrict__ src,
                                        _Float16* __restrict__ dst) {
  int i = blockIdx.x * 256 + threadIdx.x;       // i = n*256 + k, over 65536
  int n = i >> 8, k = i & 255;
  dst[i] = (_Float16)src[k * DD + n];
}

// W1 [H][64 (k)][128 (n)] f32 -> [H][n][k] f16
__global__ void cvt_transpose_W1_kernel(const float* __restrict__ src,
                                        _Float16* __restrict__ dst) {
  int i = blockIdx.x * 256 + threadIdx.x;       // over H*128*64 = 32768
  int h = i >> 13;
  int rem = i & 8191;
  int n = rem >> 6, k = rem & 63;
  dst[i] = (_Float16)src[h * (CDIM * C2) + k * C2 + n];
}

// ------------------------------------------------------ scatter-mean inputs

__global__ void count_kernel(const int* __restrict__ col,
                             float* __restrict__ cnt, int E) {
  int i = blockIdx.x * blockDim.x + threadIdx.x;
  if (i < E) atomicAdd(&cnt[col[i]], 1.0f);
}

// one block (256 thr) per edge: coalesced gather of x[row], atomic scatter
__global__ __launch_bounds__(256) void scatter_kernel(
    const float* __restrict__ x, const int* __restrict__ row,
    const int* __restrict__ col, float* hx_acc) {
  const int e = blockIdx.x;
  const int d = threadIdx.x;
  const int r = row[e];
  const int c = col[e];
  atomicAdd(&hx_acc[(size_t)c * DD + d], x[(size_t)r * DD + d]);
}

// ------------------------------------------- GEMM1: xc = x_clique + relu(hx@Wp+bp)

__global__ __launch_bounds__(512) void gemm_xc_kernel(
    const float* hx_acc, const float* __restrict__ cnt,
    const float* __restrict__ x_clique, const _Float16* __restrict__ Wp16T,
    const float* __restrict__ bp, float* xc_out) {
  __shared__ float    As32[16 * DD];               // 16 KB raw tile (async dest)
  __shared__ _Float16 As[16 * DD];                 // 8 KB f16 A tile
  const int row0 = blockIdx.x << 4;
  const int t = threadIdx.x;

  // async stage: 16 rows x 256 f32, contiguous region -> LDS (ASYNCcnt path)
  {
    const float* gbase = hx_acc + (size_t)row0 * DD;
    const unsigned lbase = lds_off(As32);
#pragma unroll
    for (int it = 0; it < 2; ++it) {
      const int idx4 = t + it * 512;               // float4 index, 1024 total
      async_g2lds_b128(lbase + idx4 * 16, gbase + idx4 * 4);
    }
    wait_asynccnt0();
  }
  __syncthreads();

  // LDS->LDS transform: scale by 1/cnt and convert to f16
  for (int j = t; j < 16 * DD; j += 512) {
    const int r = j >> 8;
    const float inv = 1.0f / fmaxf(cnt[row0 + r], 1.0f);
    As[j] = (_Float16)(As32[j] * inv);
  }
  __syncthreads();                                 // all hx reads done before xc writes

  const int wave = t >> 5;                         // 16 waves -> 16 N-tiles
  const int lane = t & 31;
  const int n0 = wave << 4;
  const int lr = lane & 15;
  const int hi = lane >> 4;
  const int koff = hi ? 8 : 0;

  v8f acc = {};
  for (int k0 = 0; k0 < DD; k0 += 32) {
    v16h a, b;
#pragma unroll
    for (int j = 0; j < 8; ++j) {
      a[j]     = As[lr * DD + k0 + koff + j];
      a[j + 8] = As[lr * DD + k0 + 16 + koff + j];
      b[j]     = Wp16T[(size_t)(n0 + lr) * DD + k0 + koff + j];
      b[j + 8] = Wp16T[(size_t)(n0 + lr) * DD + k0 + 16 + koff + j];
    }
    acc = __builtin_amdgcn_wmma_f32_16x16x32_f16(false, a, false, b,
                                                 (short)0, acc, false, false);
  }

  const int colN = n0 + lr;
  const float bias = bp[colN];
  const int rbase = hi ? 8 : 0;
#pragma unroll
  for (int r = 0; r < 8; ++r) {
    const size_t m = row0 + rbase + r;
    float v = acc[r] + bias;
    v = v > 0.0f ? v : 0.0f;
    xc_out[m * DD + colN] = x_clique[m * DD + colN] + v;
  }
}

// ------------------- GEMM2: per-head MLP, fused with W2 dot -> score [Nc,H]

__global__ __launch_bounds__(512) void gemm_score_kernel(
    const float* __restrict__ xc, const _Float16* __restrict__ W116T,
    const float* __restrict__ b1, const float* __restrict__ W2,
    const float* __restrict__ b2, float* __restrict__ score) {
  __shared__ float    As32[16 * DD];
  __shared__ _Float16 As[16 * DD];
  __shared__ float ss[16 * HEADS];
  const int row0 = blockIdx.x << 4;
  const int t = threadIdx.x;
  if (t < 16 * HEADS) ss[t] = 0.0f;

  // async stage: 16 rows x 256 f32 of xc, contiguous -> LDS
  {
    const float* gbase = xc + (size_t)row0 * DD;
    const unsigned lbase = lds_off(As32);
#pragma unroll
    for (int it = 0; it < 2; ++it) {
      const int idx4 = t + it * 512;
      async_g2lds_b128(lbase + idx4 * 16, gbase + idx4 * 4);
    }
    wait_asynccnt0();
  }
  __syncthreads();
  for (int j = t; j < 16 * DD; j += 512)
    As[j] = (_Float16)As32[j];
  __syncthreads();

  const int wave = t >> 5;
  const int lane = t & 31;
  const int h = wave >> 2;                         // head 0..3
  const int seg = wave & 3;                        // 32-col segment of 2C=128
  const int lr = lane & 15;
  const int hi = lane >> 4;
  const int koff = hi ? 8 : 0;
  const _Float16* BhT = W116T + h * (C2 * CDIM);   // [n(128)][k(64)]

  float partial[8];
#pragma unroll
  for (int r = 0; r < 8; ++r) partial[r] = 0.0f;

#pragma unroll
  for (int ct = 0; ct < 2; ++ct) {
    const int n0 = seg * 32 + ct * 16;
    v8f acc = {};
#pragma unroll
    for (int k0 = 0; k0 < CDIM; k0 += 32) {
      v16h a, b;
#pragma unroll
      for (int j = 0; j < 8; ++j) {
        a[j]     = As[lr * DD + h * CDIM + k0 + koff + j];
        a[j + 8] = As[lr * DD + h * CDIM + k0 + 16 + koff + j];
        b[j]     = BhT[(n0 + lr) * CDIM + k0 + koff + j];
        b[j + 8] = BhT[(n0 + lr) * CDIM + k0 + 16 + koff + j];
      }
      acc = __builtin_amdgcn_wmma_f32_16x16x32_f16(false, a, false, b,
                                                   (short)0, acc, false, false);
    }
    const int n = n0 + lr;
    const float w2 = W2[h * C2 + n];
    const float bb = b1[h * C2 + n];
#pragma unroll
    for (int r = 0; r < 8; ++r) {
      float v = acc[r] + bb;
      v = v > 0.0f ? v : 0.0f;                     // relu(h1)
      partial[r] += v * w2;                        // fuse W2 dot
    }
  }

  // reduce over the 16 columns held by each 16-lane half (wave32 layout)
#pragma unroll
  for (int off = 1; off < 16; off <<= 1) {
#pragma unroll
    for (int r = 0; r < 8; ++r)
      partial[r] += __shfl_xor(partial[r], off, 32);
  }
  const int rbase = hi ? 8 : 0;
  if (lr == 0) {
#pragma unroll
    for (int r = 0; r < 8; ++r)
      atomicAdd(&ss[(rbase + r) * HEADS + h], partial[r]);
  }
  __syncthreads();
  if (t < 16 * HEADS) {
    const int r = t >> 2;
    const int hh = t & 3;
    score[(size_t)(row0 + r) * HEADS + hh] = ss[t] + b2[hh];
  }
}

// ---------------------------------------------------- segment softmax + pool

__device__ __forceinline__ unsigned fkey(float f) {
  unsigned b = __float_as_uint(f);
  return (b & 0x80000000u) ? ~b : (b | 0x80000000u);
}
__device__ __forceinline__ float fdecode(unsigned u) {
  return (u & 0x80000000u) ? __uint_as_float(u & 0x7FFFFFFFu)
                           : __uint_as_float(~u);
}

__global__ void segmax_kernel(const float* __restrict__ score,
                              const int* __restrict__ batch,
                              unsigned* __restrict__ mxkey, int Nc) {
  int i = blockIdx.x * blockDim.x + threadIdx.x;
  if (i < Nc * HEADS) {
    int n = i >> 2, h = i & 3;
    atomicMax(&mxkey[batch[n] * HEADS + h], fkey(score[i]));
  }
}

__global__ void denom_kernel(float* e_io /* score in, e out */,
                             const int* __restrict__ batch,
                             const unsigned* __restrict__ mxkey,
                             float* __restrict__ denom, int Nc) {
  int i = blockIdx.x * blockDim.x + threadIdx.x;
  if (i < Nc * HEADS) {
    int n = i >> 2, h = i & 3;
    unsigned k = mxkey[batch[n] * HEADS + h];
    float mx = (k == 0u) ? 0.0f : fdecode(k);
    float e = __expf(e_io[i] - mx);
    e_io[i] = e;
    atomicAdd(&denom[batch[n] * HEADS + h], e);
  }
}

// one block per clique (256 thr = D): write alpha, scatter sc*alpha into drug_feat
__global__ __launch_bounds__(256) void finalize_kernel(
    float* alpha_io /* e in, alpha out */, const float* __restrict__ xc,
    const int* __restrict__ batch, const float* __restrict__ denom,
    float* __restrict__ drug) {
  __shared__ float a_sh[HEADS];
  const int n = blockIdx.x;
  const int d = threadIdx.x;
  const int b = batch[n];
  if (d < HEADS) {
    float a = alpha_io[(size_t)n * HEADS + d] /
              (denom[b * HEADS + d] + 1e-16f);
    alpha_io[(size_t)n * HEADS + d] = a;
    a_sh[d] = a;
  }
  __syncthreads();
  const float v = xc[(size_t)n * DD + d] * a_sh[d >> 6];   // head = d / C
  atomicAdd(&drug[(size_t)b * DD + d], v);
}

// ---------------------------------------------------------------- launcher

extern "C" void kernel_launch(void* const* d_in, const int* in_sizes, int n_in,
                              void* d_out, int out_size, void* d_ws, size_t ws_size,
                              hipStream_t stream) {
  const float* x        = (const float*)d_in[0];
  const float* x_clique = (const float*)d_in[1];
  const float* Wp = (const float*)d_in[2];
  const float* bp = (const float*)d_in[3];
  const float* W1 = (const float*)d_in[4];
  const float* b1 = (const float*)d_in[5];
  const float* W2 = (const float*)d_in[6];
  const float* b2 = (const float*)d_in[7];
  const int* a2c    = (const int*)d_in[8];   // [2, E]
  const int* cbatch = (const int*)d_in[9];   // [Nc]
  // d_in[10] clique_edge_index: unused by the reference

  const int E  = in_sizes[8] / 2;
  const int Nc = in_sizes[9];

  float* out   = (float*)d_out;
  float* drug  = out;                                  // [B, 256]
  float* xcreg = out + (size_t)NGRAPH * DD;            // [Nc, 256] (also hx_sum accumulator)
  float* alpha = xcreg + (size_t)Nc * DD;              // [Nc, 4]   (also score / e buffer)

  char* w = (char*)d_ws;
  float*    cnt   = (float*)w;    w += (size_t)Nc * sizeof(float);
  unsigned* mxkey = (unsigned*)w; w += (size_t)NGRAPH * HEADS * sizeof(unsigned);
  float*    denom = (float*)w;    w += (size_t)NGRAPH * HEADS * sizeof(float);
  _Float16* Wp16T = (_Float16*)w; w += (size_t)DD * DD * sizeof(_Float16);
  _Float16* W116T = (_Float16*)w;

  // init accumulators
  zero_f32_kernel<<<2048, 256, 0, stream>>>(drug, (long long)NGRAPH * DD);
  zero_f32_kernel<<<8192, 256, 0, stream>>>(xcreg, (long long)Nc * DD);
  zero_f32_kernel<<<1024, 256, 0, stream>>>(cnt, Nc);
  zero_f32_kernel<<<32, 256, 0, stream>>>(denom, NGRAPH * HEADS);
  zero_f32_kernel<<<32, 256, 0, stream>>>((float*)mxkey, NGRAPH * HEADS);

  // f16 transposed weights
  cvt_transpose_Wp_kernel<<<DD * DD / 256, 256, 0, stream>>>(Wp, Wp16T);
  cvt_transpose_W1_kernel<<<HEADS * CDIM * C2 / 256, 256, 0, stream>>>(W1, W116T);

  // scatter-mean of atoms into cliques
  count_kernel<<<(E + 255) / 256, 256, 0, stream>>>(a2c + E, cnt, E);
  scatter_kernel<<<E, 256, 0, stream>>>(x, a2c, a2c + E, xcreg);

  // xc = x_clique + relu(hx @ Wp + bp)   (WMMA, async-LDS staged, in-place)
  gemm_xc_kernel<<<Nc / 16, 512, 0, stream>>>(xcreg, cnt, x_clique, Wp16T, bp, xcreg);

  // per-head MLP -> score (WMMA, fused relu + W2 dot)
  gemm_score_kernel<<<Nc / 16, 512, 0, stream>>>(xcreg, W116T, b1, W2, b2, alpha);

  // segment softmax per (graph, head)
  segmax_kernel<<<(Nc * HEADS + 255) / 256, 256, 0, stream>>>(alpha, cbatch, mxkey, Nc);
  denom_kernel<<<(Nc * HEADS + 255) / 256, 256, 0, stream>>>(alpha, cbatch, mxkey, denom, Nc);

  // alpha + weighted segment-sum pooling
  finalize_kernel<<<Nc, 256, 0, stream>>>(alpha, xcreg, cbatch, denom, drug);
}
